// FourierGrid_79766132621757
// MI455X (gfx1250) — matrix operand order repaired and checked by reading, not compiled, not run on hardware
//
#include <hip/hip_runtime.h>

// ---------------------------------------------------------------------------
// FourierGrid Gram matrix:  C = J @ J^T,  J: [8192 x 4096] sparse (<=16 nnz/row)
// Densify J in f16 (64 MB -> fits MI455X 192MB L2), then tiled WMMA GEMM with
// double-buffered LDS and gfx1250 async global->LDS copies (ASYNCcnt path).
// ---------------------------------------------------------------------------

typedef _Float16 half8  __attribute__((ext_vector_type(8)));
typedef _Float16 half16 __attribute__((ext_vector_type(16)));
typedef float    float8 __attribute__((ext_vector_type(8)));
typedef int      i32x4  __attribute__((ext_vector_type(4)));

#define NPTS     8192
#define KDIM     4096
#define GRIDLEN  512
#define BANDS    8
#define LDS_STR  40   // halves per LDS row (32 data + 8 pad to spread banks)

#if defined(__AMDGCN__) && __has_builtin(__builtin_amdgcn_global_load_async_to_lds_b128)
#define USE_ASYNC 1
#else
#define USE_ASYNC 0
#endif

#if USE_ASYNC
typedef __attribute__((address_space(1))) i32x4 gas_i32x4;   // global b128
typedef __attribute__((address_space(3))) i32x4 las_i32x4;   // LDS b128
#if __has_builtin(__builtin_amdgcn_s_wait_asynccnt)
#define WAIT_ASYNC0() __builtin_amdgcn_s_wait_asynccnt(0)
#else
#define WAIT_ASYNC0() asm volatile("s_wait_asynccnt 0" ::: "memory")
#endif
// Copy this thread's 32B slot (16 halves) global -> LDS, two async b128 ops.
// ISA: IOFFSET is added to BOTH the global and LDS addresses.
// Global AS1 cast goes through an integer (flat==global address values);
// LDS AS3 cast is a real addrspacecast from the flat __shared__ pointer.
#define ASYNC_CP32(gp, lp)                                                   \
    do {                                                                     \
        gas_i32x4* _g = (gas_i32x4*)(unsigned long long)(gp);                \
        las_i32x4* _l = (las_i32x4*)(lp);                                    \
        __builtin_amdgcn_global_load_async_to_lds_b128(_g, _l, 0, 0);        \
        __builtin_amdgcn_global_load_async_to_lds_b128(_g, _l, 16, 0);       \
    } while (0)
#endif

// ---------------------------------------------------------------------------
// Kernel 1: zero the J workspace (64 MB) with 16-byte stores.
// ---------------------------------------------------------------------------
__global__ void fg_zero_ws(uint4* __restrict__ p, int n16) {
    uint4 z; z.x = 0u; z.y = 0u; z.z = 0u; z.w = 0u;
    for (int i = blockIdx.x * blockDim.x + threadIdx.x; i < n16;
         i += gridDim.x * blockDim.x) {
        p[i] = z;
    }
}

// ---------------------------------------------------------------------------
// Kernel 2: scatter Fourier-grid interpolation weights into dense f16 J.
// One thread per (point, band); reproduces the reference exactly, including
// the Python `2 ^ (i // 2)` bitwise-XOR frequency quirk.
// ---------------------------------------------------------------------------
__global__ void fg_build_J(_Float16* __restrict__ J) {
    int t = blockIdx.x * blockDim.x + threadIdx.x;
    if (t >= NPTS * BANDS) return;
    int n = t >> 3;
    int b = t & 7;

    float x    = (float)n * (1.0f / (float)NPTS);
    int   freq = 2 ^ (b >> 1);                   // 2,2,3,3,0,0,1,1
    float ph   = (float)freq * 3.14159265358979323846f * x;
    float raw  = (b & 1) ? cosf(ph) : sinf(ph);
    float fr   = (raw + 1.0f) * 0.5f;            // in [0,1]

    float lf = floorf(fr * (float)GRIDLEN);
    float lw = fabsf(fr - (lf + 1.0f) * (1.0f / (float)GRIDLEN)) * (float)GRIDLEN;
    float rw = fabsf(fr - lf          * (1.0f / (float)GRIDLEN)) * (float)GRIDLEN;
    int li = (int)lf;
    int ri = li + 1;

    size_t rowBase = (size_t)n * KDIM + (size_t)b * GRIDLEN;
    if (li > 0 && li < GRIDLEN) J[rowBase + li] = (_Float16)lw;
    if (ri < GRIDLEN)           J[rowBase + ri] = (_Float16)rw;
}

// ---------------------------------------------------------------------------
// Per-slab compute: load fragments from one LDS buffer, do 4x2 WMMAs.
// A 16x32 f16 layout: lane row M=lane&15; elems 0-7 -> K=kbase..+7,
// elems 8-15 -> K=kbase+16..+23, kbase = 8*(lane>>4).
// B 32x16 f16 layout: lane col N=lane&15; same K mapping as A:
// elems 0-7 -> K=kbase..+7, elems 8-15 -> K=kbase+16..+23.
// ---------------------------------------------------------------------------
__device__ __forceinline__ void fg_step(const _Float16* AsBuf,
                                        const _Float16* BsBuf,
                                        int wm, int wn, int laneR, int laneHi,
                                        float8 acc[4][2]) {
    half16 af[4];
#pragma unroll
    for (int mt = 0; mt < 4; ++mt) {
        const _Float16* p = AsBuf + (wm * 64 + mt * 16 + laneR) * LDS_STR + laneHi * 8;
        half8 lo = *(const half8*)(p);
        half8 hi = *(const half8*)(p + 16);
        af[mt] = __builtin_shufflevector(lo, hi, 0,1,2,3,4,5,6,7,
                                                 8,9,10,11,12,13,14,15);
    }
    half16 bf[2];
#pragma unroll
    for (int nt = 0; nt < 2; ++nt) {
        const _Float16* p = BsBuf + (wn * 32 + nt * 16 + laneR) * LDS_STR + laneHi * 8;
        half8 lo = *(const half8*)(p);
        half8 hi = *(const half8*)(p + 16);
        bf[nt] = __builtin_shufflevector(lo, hi, 0,1,2,3,4,5,6,7,
                                                 8,9,10,11,12,13,14,15);
    }
#pragma unroll
    for (int mt = 0; mt < 4; ++mt)
#pragma unroll
        for (int nt = 0; nt < 2; ++nt)
            acc[mt][nt] = __builtin_amdgcn_wmma_f32_16x16x32_f16(
                false, af[mt], false, bf[nt], (short)0, acc[mt][nt],
                false, false);
}

// ---------------------------------------------------------------------------
// Kernel 3: C = J @ J^T. Block: 256 threads = 8 wave32s -> 128x128 tile.
// Wave (wm,wn): 64x32 sub-tile = 4x2 accumulators. K staged in 32-wide slabs,
// double-buffered LDS, async global->LDS prefetch one slab ahead.
// ---------------------------------------------------------------------------
__global__ void __launch_bounds__(256)
fg_gram_wmma(const _Float16* __restrict__ J, float* __restrict__ C) {
    __shared__ _Float16 As[2][128 * LDS_STR];
    __shared__ _Float16 Bs[2][128 * LDS_STR];

    const int tid    = threadIdx.x;
    const int lane   = tid & 31;
    const int wave   = tid >> 5;
    const int wm     = wave & 1;        // M strip within block (x64)
    const int wn     = wave >> 1;       // N strip within block (x32)
    const int laneR  = lane & 15;
    const int laneHi = lane >> 4;

    const int mBlock = blockIdx.y * 128;
    const int nBlock = blockIdx.x * 128;

    // Staging: thread t owns 16 contiguous halves (32B) of one row per slab.
    const int srow  = tid >> 1;         // 0..127
    const int shalf = (tid & 1) * 16;   // half-element offset 0 or 16

    const _Float16* gA = J + (size_t)(mBlock + srow) * KDIM + shalf;
    const _Float16* gB = J + (size_t)(nBlock + srow) * KDIM + shalf;
    const int sOff = srow * LDS_STR + shalf;

    float8 acc[4][2] = {};

#if USE_ASYNC
    // Prefetch slab 0 into buffer 0.
    ASYNC_CP32(gA, &As[0][sOff]);
    ASYNC_CP32(gB, &Bs[0][sOff]);
    WAIT_ASYNC0();
    __syncthreads();

    for (int k0 = 0; k0 < KDIM; k0 += 64) {
        // --- slab k0 in buf0; prefetch slab k0+32 into buf1 ---
        ASYNC_CP32(gA + k0 + 32, &As[1][sOff]);
        ASYNC_CP32(gB + k0 + 32, &Bs[1][sOff]);
        fg_step(&As[0][0], &Bs[0][0], wm, wn, laneR, laneHi, acc);
        WAIT_ASYNC0();
        __syncthreads();

        // --- slab k0+32 in buf1; prefetch slab k0+64 into buf0 ---
        if (k0 + 64 < KDIM) {
            ASYNC_CP32(gA + k0 + 64, &As[0][sOff]);
            ASYNC_CP32(gB + k0 + 64, &Bs[0][sOff]);
        }
        fg_step(&As[1][0], &Bs[1][0], wm, wn, laneR, laneHi, acc);
        WAIT_ASYNC0();
        __syncthreads();
    }
#else
    // Fallback: single-buffer staging through VGPRs (proven path).
    _Float16* sA = &As[0][sOff];
    _Float16* sB = &Bs[0][sOff];
    for (int k0 = 0; k0 < KDIM; k0 += 32) {
        __syncthreads();
        uint4 a0 = ((const uint4*)(gA + k0))[0];
        uint4 a1 = ((const uint4*)(gA + k0))[1];
        uint4 b0 = ((const uint4*)(gB + k0))[0];
        uint4 b1 = ((const uint4*)(gB + k0))[1];
        ((uint4*)sA)[0] = a0;
        ((uint4*)sA)[1] = a1;
        ((uint4*)sB)[0] = b0;
        ((uint4*)sB)[1] = b1;
        __syncthreads();
        fg_step(&As[0][0], &Bs[0][0], wm, wn, laneR, laneHi, acc);
    }
#endif

    // C/D layout: VGPR j -> row M = j + 8*laneHi, column N = lane&15.
#pragma unroll
    for (int mt = 0; mt < 4; ++mt) {
#pragma unroll
        for (int nt = 0; nt < 2; ++nt) {
            const int col = nBlock + wn * 32 + nt * 16 + laneR;
#pragma unroll
            for (int j = 0; j < 8; ++j) {
                const int row = mBlock + wm * 64 + mt * 16 + laneHi * 8 + j;
                C[(size_t)row * NPTS + col] = acc[mt][nt][j];
            }
        }
    }
}

// ---------------------------------------------------------------------------
extern "C" void kernel_launch(void* const* d_in, const int* in_sizes, int n_in,
                              void* d_out, int out_size, void* d_ws, size_t ws_size,
                              hipStream_t stream) {
    (void)d_in; (void)in_sizes; (void)n_in; (void)out_size; (void)ws_size;

    _Float16* J = (_Float16*)d_ws;               // 8192*4096*2 = 64 MB
    float*    C = (float*)d_out;                 // 8192*8192*4 = 256 MB

    const int n16 = (int)(((size_t)NPTS * KDIM * sizeof(_Float16)) / 16);
    fg_zero_ws<<<4096, 256, 0, stream>>>((uint4*)d_ws, n16);

    fg_build_J<<<(NPTS * BANDS + 255) / 256, 256, 0, stream>>>(J);

    dim3 grid(NPTS / 128, NPTS / 128);
    fg_gram_wmma<<<grid, 256, 0, stream>>>(J, C);
}
